// LSTM_3designs_12567074308218
// MI455X (gfx1250) — matrix operand order repaired and checked by reading, not compiled, not run on hardware
//
#include <hip/hip_runtime.h>

typedef __attribute__((ext_vector_type(16))) __bf16 v16bf;
typedef __attribute__((ext_vector_type(8)))  float  v8f;

// ---------- helpers ----------
__device__ __forceinline__ unsigned short f2bf(float f) {
  unsigned int u = __builtin_bit_cast(unsigned int, f);
  u += 0x7fffu + ((u >> 16) & 1u);          // round-to-nearest-even
  return (unsigned short)(u >> 16);
}

// Index (in ushorts) of element (chunk c, k-in-chunk kk, row m) inside an LDS
// buffer holding 16x32 bf16 A-fragments in the exact v_wmma A-operand layout:
//   lanes 0-15 : halfs 0-7 = K 0-7,  halfs 8-15 = K 16-23   (M = lane)
//   lanes 16-31: halfs 0-7 = K 8-15, halfs 8-15 = K 24-31   (M = lane-16)
__device__ __forceinline__ int a_idx(int c, int kk, int m) {
  int hi = (kk >> 3) & 1;
  int j  = (kk & 7) + ((kk & 16) ? 8 : 0);
  return ((c * 32) + (hi * 16 + m)) * 16 + j;
}

__device__ __forceinline__ float sigm(float x) { return 1.f / (1.f + __expf(-x)); }

// ---------- weight repack: [Wih;Whh]^T (f32) -> bf16 B-fragment layout ----------
// B operand layout (32x16, K x N): lane = 16*hi + n_local,
// halfs 0-7 = K (hi*8 .. hi*8+7), halfs 8-15 = K (16+hi*8 .. 16+hi*8+7) for column n.
__global__ void pack_weights(const float* __restrict__ Wih,
                             const float* __restrict__ Whh,
                             int IN, int H, unsigned short* __restrict__ out) {
  const int  NG = 4 * H;
  const int  K  = IN + H;
  const long total = (long)K * NG;
  long tid = (long)blockIdx.x * blockDim.x + threadIdx.x;
  if (tid >= total) return;
  int k = (int)(tid / NG);
  int n = (int)(tid % NG);
  float v = (k < IN) ? Wih[(long)n * IN + k] : Whh[(long)n * H + (k - IN)];
  int c = k >> 5, kc = k & 31;
  int hi = (kc >> 3) & 1, j = (kc & 7) + ((kc & 16) ? 8 : 0);
  int nt = n >> 4, nl = n & 15;
  int NT = NG >> 4;
  long addr = (((long)c * NT + nt) * 32 + (hi * 16 + nl)) * 16 + j;
  out[addr] = f2bf(v);
}

// ---------- fused LSTM recurrence on WMMA ----------
// One block = 16 batch rows, all T timesteps. 8 waves; wave w owns hidden slice
// [w*H/8, (w+1)*H/8) for ALL four gates, so the i/f/g/o values for a unit live
// at identical (lane, vgpr) positions of the wave's 4 accumulator groups.
// A fragments ([x_t|h] tile) are held in registers for the whole step; weight
// B fragments are streamed from L2 every step through an LICM-opaque offset
// (offset-based so the pointer keeps global addrspace -> global_load_b128,
// pure LOADcnt, no FLAT/DScnt coupling).
template <int IN, int H>
__global__ __launch_bounds__(256)
void lstm_wmma(const float* __restrict__ x,            // [NB, T, IN] f32
               const unsigned short* __restrict__ Wp,  // packed bf16 [K/32][NT][32][16]
               const float* __restrict__ bih,
               const float* __restrict__ bhh,
               float* __restrict__ feat,               // [B, 3840] f32
               int T, int colBase, int P) {
  constexpr int NG  = 4 * H;
  constexpr int KCH = (IN + H) / 32;   // 32-wide K chunks
  constexpr int NT  = NG / 16;         // total 16-col N tiles
  constexpr int HPW = H / 8;           // hidden units per wave
  constexpr int TPG = HPW / 16;        // N tiles per gate per wave
  constexpr int FLD = 3840;            // feature-buffer leading dim

  __shared__ __align__(32) unsigned short Abuf[KCH * 512]; // [x_t | h_{t-1}] bf16, A layout
  __shared__ float cbuf[16 * H];                           // cell state f32

  const int tid  = threadIdx.x;
  const int wv   = tid >> 5;
  const int lane = tid & 31;
  const int b0   = blockIdx.x * 16;
  const int mh   = lane >> 4;       // C/D layout: m = r + 8*mh
  const int nl   = lane & 15;       //             n = nl

  for (int i = tid; i < KCH * 512; i += 256) Abuf[i] = 0;       // h_{-1} = 0
  for (int i = tid; i < 16 * H;    i += 256) cbuf[i] = 0.f;     // c_{-1} = 0
  __syncthreads();

  v8f acc[4][TPG];

  for (int t = 0; t < T; ++t) {
    // ---- stage x_t tile (16 x IN) into Abuf, f32 -> bf16, A-fragment layout ----
    for (int e = tid; e < 16 * IN; e += 256) {
      int m = e / IN, k = e % IN;
      float v = x[((long)(b0 + m) * T + t) * IN + k];
      Abuf[a_idx(k >> 5, k & 31, m)] = f2bf(v);
    }
    __syncthreads();   // x writes + previous h writes visible to all waves

    // ---- load all A fragments for this step into registers (reused 4*TPG times)
    v16bf afrag[KCH];
    #pragma unroll
    for (int c = 0; c < KCH; ++c)
      afrag[c] = *reinterpret_cast<const v16bf*>(&Abuf[(c * 32 + lane) * 16]);

    // LICM-opaque *offset* (pointer provenance preserved => global_load_b128):
    // forces B fragments to stream from L2 each step instead of being hoisted
    // out of the t-loop and spilled to scratch.
    unsigned int wofs = 0;
    asm volatile("" : "+s"(wofs));
    const unsigned short* wp = Wp + wofs;

    // ---- gates = [x_t | h_{t-1}] @ [Wih;Whh]^T via bf16 WMMA, f32 accumulate ----
    #pragma unroll
    for (int gi = 0; gi < 4; ++gi) {
      #pragma unroll
      for (int tt = 0; tt < TPG; ++tt) {
        const int nt = gi * (H / 16) + wv * TPG + tt;
        v8f a0 = {};
        acc[gi][tt] = a0;
        #pragma unroll
        for (int c = 0; c < KCH; ++c) {
          v16bf b = *reinterpret_cast<const v16bf*>(
              &wp[(((long)c * NT + nt) * 32 + lane) * 16]);
          acc[gi][tt] = __builtin_amdgcn_wmma_f32_16x16x32_bf16(
              false, afrag[c], false, b, (short)0, acc[gi][tt], false, false);
        }
      }
    }
    __syncthreads();   // all waves done reading Abuf h-region before overwrite

    // ---- activations + state update, entirely in registers per lane ----
    #pragma unroll
    for (int tt = 0; tt < TPG; ++tt) {
      const int j = wv * HPW + tt * 16 + nl;
      const float bi = bih[0 * H + j] + bhh[0 * H + j];
      const float bf = bih[1 * H + j] + bhh[1 * H + j];
      const float bg = bih[2 * H + j] + bhh[2 * H + j];
      const float bo = bih[3 * H + j] + bhh[3 * H + j];
      #pragma unroll
      for (int r = 0; r < 8; ++r) {
        const int m = r + 8 * mh;
        float gi_ = acc[0][tt][r] + bi;
        float gf_ = acc[1][tt][r] + bf;
        float gg_ = acc[2][tt][r] + bg;
        float go_ = acc[3][tt][r] + bo;
        float cold = cbuf[m * H + j];
        float cn = sigm(gf_) * cold + sigm(gi_) * tanhf(gg_);
        float h  = sigm(go_) * tanhf(cn);
        cbuf[m * H + j] = cn;
        const int k = IN + j;
        Abuf[a_idx(k >> 5, k & 31, m)] = f2bf(h);
        if (t == T - 1) {
          const int row = b0 + m;
          if (P > 0)
            feat[(long)(row / P) * FLD + colBase + (row % P) * H + j] = h;
          else
            feat[(long)row * FLD + colBase + j] = h;
        }
      }
    }
    // next iteration's stage-x barrier also covers these h/c writes
  }
}

// ---------- FC head: out[b,o] = feat[b,:] . Wfc[o,:3840] + cg[b,:] . Wfc[o,3840:] + bfc[o] ----------
__global__ void fc_kernel(const float* __restrict__ feat, const float* __restrict__ cg,
                          const float* __restrict__ Wfc,  const float* __restrict__ bfc,
                          float* __restrict__ out) {
  const int b = blockIdx.x;
  const int tid = threadIdx.x;                 // 128 threads
  float s0 = 0.f, s1 = 0.f;
  for (int col = tid; col < 3840; col += 128) {
    float v = feat[(long)b * 3840 + col];
    s0 += v * Wfc[col];
    s1 += v * Wfc[3872 + col];
  }
  if (tid < 32) {
    float v = cg[b * 32 + tid];
    s0 += v * Wfc[3840 + tid];
    s1 += v * Wfc[3872 + 3840 + tid];
  }
  __shared__ float red[256];
  red[tid] = s0;
  red[128 + tid] = s1;
  __syncthreads();
  for (int off = 64; off > 0; off >>= 1) {
    if (tid < off) {
      red[tid]       += red[tid + off];
      red[128 + tid] += red[128 + tid + off];
    }
    __syncthreads();
  }
  if (tid == 0) {
    out[b * 2 + 0] = red[0]   + bfc[0];
    out[b * 2 + 1] = red[128] + bfc[1];
  }
}

// ---------- launch ----------
extern "C" void kernel_launch(void* const* d_in, const int* in_sizes, int n_in,
                              void* d_out, int out_size, void* d_ws, size_t ws_size,
                              hipStream_t stream) {
  (void)in_sizes; (void)n_in; (void)out_size; (void)ws_size;

  const float* cg   = (const float*)d_in[0];
  const float* gh   = (const float*)d_in[1];
  const float* ga   = (const float*)d_in[2];
  const float* gv   = (const float*)d_in[3];
  const float* ph   = (const float*)d_in[4];
  const float* pa   = (const float*)d_in[5];
  const float* Wih1 = (const float*)d_in[6];
  const float* Whh1 = (const float*)d_in[7];
  const float* bih1 = (const float*)d_in[8];
  const float* bhh1 = (const float*)d_in[9];
  const float* Wih2 = (const float*)d_in[10];
  const float* Whh2 = (const float*)d_in[11];
  const float* bih2 = (const float*)d_in[12];
  const float* bhh2 = (const float*)d_in[13];
  const float* Wih3 = (const float*)d_in[14];
  const float* Whh3 = (const float*)d_in[15];
  const float* bih3 = (const float*)d_in[16];
  const float* bhh3 = (const float*)d_in[17];
  const float* Wfc  = (const float*)d_in[18];
  const float* bfc  = (const float*)d_in[19];

  // workspace layout (bytes): Wp1 655360 | Wp2 655360 | Wp3 163840 | feat 512*3840*4
  char* ws = (char*)d_ws;
  unsigned short* Wp1 = (unsigned short*)(ws);
  unsigned short* Wp2 = (unsigned short*)(ws + 655360);
  unsigned short* Wp3 = (unsigned short*)(ws + 655360 * 2);
  float*          feat = (float*)(ws + 655360 * 2 + 163840);

  // one-time bf16 repack of [Wih;Whh]^T into WMMA B-fragment order
  pack_weights<<<1280, 256, 0, stream>>>(Wih1, Whh1, 64, 256, Wp1);  // 320*1024
  pack_weights<<<1280, 256, 0, stream>>>(Wih2, Whh2, 64, 256, Wp2);  // 320*1024
  pack_weights<<< 320, 256, 0, stream>>>(Wih3, Whh3, 32, 128, Wp3);  // 160*512

  // LSTMs: feat columns = [home 0..255 | away 256..511 | vs 512..767 |
  //                        players_home 768..2303 | players_away 2304..3839]
  lstm_wmma<64, 256><<< 32, 256, 0, stream>>>(gh, Wp1, bih1, bhh1, feat, 128,    0,  0);
  lstm_wmma<64, 256><<< 32, 256, 0, stream>>>(ga, Wp1, bih1, bhh1, feat, 128,  256,  0);
  lstm_wmma<64, 256><<< 32, 256, 0, stream>>>(gv, Wp2, bih2, bhh2, feat, 128,  512,  0);
  lstm_wmma<32, 128><<<384, 256, 0, stream>>>(ph, Wp3, bih3, bhh3, feat,  64,  768, 12);
  lstm_wmma<32, 128><<<384, 256, 0, stream>>>(pa, Wp3, bih3, bhh3, feat,  64, 2304, 12);

  fc_kernel<<<512, 128, 0, stream>>>(feat, cg, Wfc, bfc, (float*)d_out);
}
